// Attention_22050362097758
// MI455X (gfx1250) — matrix operand order repaired and checked by reading, not compiled, not run on hardware
//
#include <hip/hip_runtime.h>
#include <math.h>
#include <stdint.h>

// ---------------------------------------------------------------- types ----
typedef __bf16 bf16_t;
typedef __attribute__((ext_vector_type(16))) __bf16 v16bf;
typedef __attribute__((ext_vector_type(8)))  float  v8f;

#define NEG_MAX  (-3.402823466e38f)

// D = A(16x32 bf16) * B(32x16 bf16) + C(16x16 f32), wave32 WMMA
static __device__ inline v8f wmma_bf16(v16bf a, v16bf b, v8f c) {
  return __builtin_amdgcn_wmma_f32_16x16x32_bf16(
      /*neg_a=*/false, a, /*neg_b=*/false, b,
      /*c_mod=*/(short)0, c, /*reuse_a=*/false, /*reuse_b=*/false);
}

// A-operand: 16x32 tile of a row-major [M x K] bf16 matrix, origin p0, ld elems.
// ISA layout: lane m(0..15), half=lane>>4: VGPR v<4 -> K=2v+8h ; v>=4 -> K=16+2(v-4)+8h
static __device__ inline v16bf load_a_tile(const bf16_t* p0, int ld) {
  int lane = threadIdx.x & 31;
  const bf16_t* p = p0 + (lane & 15) * ld + (lane >> 4) * 8;
  p = (const bf16_t*)__builtin_assume_aligned(p, 16);
  v16bf a;
  __builtin_memcpy(&a, p, 16);                        // elements 0..7  (K = 8h+0..7)
  __builtin_memcpy((char*)&a + 16, p + 16, 16);       // elements 8..15 (K = 8h+16..23)
  return a;
}

// B-operand: 32x16 tile where B[k][n] = src[n*ld + k] (i.e. src is N x K row-major).
// ISA layout: lane n(0..15), half: elements 0..7 -> K = 8h+0..7, elements 8..15 -> K = 16+8h+0..7.
static __device__ inline v16bf load_bt_tile(const bf16_t* p0, int ld) {
  int lane = threadIdx.x & 31;
  const bf16_t* p = p0 + (lane & 15) * ld + (lane >> 4) * 8;
  p = (const bf16_t*)__builtin_assume_aligned(p, 16);
  v16bf b;
  __builtin_memcpy(&b, p, 16);
  __builtin_memcpy((char*)&b + 16, p + 16, 16);
  return b;
}

// ------------------------------------------------------------ prep kernels ----
// RMSNorm over last dim (256) of x [1024 rows], apply gamma, write bf16.
__global__ __launch_bounds__(256) void rmsnorm_x_kernel(const float* __restrict__ x,
                                                        const float* __restrict__ gamma,
                                                        bf16_t* __restrict__ xn) {
  __shared__ float red[256];
  int row = blockIdx.x, c = threadIdx.x;
  float v = x[row * 256 + c];
  red[c] = v * v;
  __syncthreads();
  for (int s = 128; s > 0; s >>= 1) {
    if (c < s) red[c] += red[c + s];
    __syncthreads();
  }
  float scale = rsqrtf(red[0] * (1.0f / 256.0f) + 1e-5f);
  xn[row * 256 + c] = (bf16_t)(v * scale * gamma[c]);
}

// dst[c][r] = (bf16) src[r][c]   (src rows x cols  ->  dst cols x rows)
__global__ __launch_bounds__(256) void transpose_to_bf16(const float* __restrict__ src,
                                                         bf16_t* __restrict__ dst,
                                                         int rows, int cols) {
  int idx = blockIdx.x * blockDim.x + threadIdx.x;
  if (idx < rows * cols) {
    int r = idx / cols, c = idx - r * cols;
    dst[c * rows + r] = (bf16_t)src[r * cols + c];
  }
}

// gwT[n][k] = (n<8) ? gamma_e[k] * W_edge[k][n] : 0   -- B-operand for edge bias GEMM
__global__ __launch_bounds__(256) void prep_gwT_kernel(const float* __restrict__ gamma_e,
                                                       const float* __restrict__ W_edge,
                                                       bf16_t* __restrict__ gwT) {
  int idx = blockIdx.x * blockDim.x + threadIdx.x;   // 16*256
  int n = idx >> 8, k = idx & 255;
  float v = (n < 8) ? gamma_e[k] * W_edge[k * 8 + n] : 0.f;
  gwT[n * 256 + k] = (bf16_t)v;
}

// ------------------------------------------------------------ QKV GEMM ----
// xn [1024 x 256] bf16  @  W_qkv -> q [bh][n][d]*8, k [bh][n][d], vT [bh][d][n]
// wqkvT is [1536 x 256] (N x K). One wave per 16x16 tile; 64x96 tiles.
__global__ __launch_bounds__(256) void qkv_gemm_kernel(const bf16_t* __restrict__ xn,
                                                       const bf16_t* __restrict__ wqkvT,
                                                       bf16_t* __restrict__ q_ws,
                                                       bf16_t* __restrict__ k_ws,
                                                       bf16_t* __restrict__ vt_ws) {
  int wave = (blockIdx.x * blockDim.x + threadIdx.x) >> 5;
  int mt = wave / 96, nt = wave - mt * 96;
  v8f acc = {0.f, 0.f, 0.f, 0.f, 0.f, 0.f, 0.f, 0.f};
#pragma unroll
  for (int k0 = 0; k0 < 256; k0 += 32)
    acc = wmma_bf16(load_a_tile(xn + mt * 16 * 256 + k0, 256),
                    load_bt_tile(wqkvT + nt * 16 * 256 + k0, 256), acc);

  int lane = threadIdx.x & 31, n16 = lane & 15, half = lane >> 4;
  int col0 = nt * 16;
  int t = col0 >> 9;               // 0=q 1=k 2=v (uniform per wave)
  int h = (col0 >> 6) & 7;         // head (uniform per wave)
  int d = (col0 & 63) + n16;
#pragma unroll
  for (int r = 0; r < 8; ++r) {
    int row = mt * 16 + r + 8 * half;
    int bidx = row >> 9, i = row & 511;
    int bh = bidx * 8 + h;
    float val = acc[r];
    if (t == 0)       q_ws[(bh * 512 + i) * 64 + d] = (bf16_t)(val * 8.0f);  // * sqrt(64)
    else if (t == 1)  k_ws[(bh * 512 + i) * 64 + d] = (bf16_t)val;
    else              vt_ws[(bh * 64 + d) * 512 + i] = (bf16_t)val;          // transposed
  }
}

// ------------------------------------------------------------ edge bias ----
// bias[b,h,i,j] = rmsnorm(edges[b,i,j,:], gamma_e) @ W_edge[:,h] + b_edge[h]
// Lower triangle only (j <= i).  WMMA formulation: A = 16 consecutive edge rows
// (contiguous [16 x 256] f32 block, converted bf16 in-register), B = gwT (regs).
// Sum(e^2) per row computed from the same f32 values; 1 shuffle per row.
__global__ __launch_bounds__(256) void edge_bias_kernel(const float* __restrict__ edges,
                                                        const bf16_t* __restrict__ gwT,
                                                        const float* __restrict__ b_edge,
                                                        float* __restrict__ bias_ws) {
  int bidx = blockIdx.x >> 9;
  int i = blockIdx.x & 511;
  int wavein = threadIdx.x >> 5;
  int lane = threadIdx.x & 31, n16 = lane & 15, half = lane >> 4;

  // B-operand tiles (head-projection matrix), resident in VGPRs for whole kernel
  v16bf btile[8];
#pragma unroll
  for (int t = 0; t < 8; ++t) btile[t] = load_bt_tile(gwT + t * 32, 256);
  float beL = b_edge[n16 & 7];

  const float* erow = edges + ((size_t)(bidx * 512 + i)) * 512 * 256;
  float* bias_bi = bias_ws + ((size_t)(bidx * 8) * 512 + i) * 512;

  for (int jt = wavein; jt * 16 <= i; jt += 8) {
    int j0 = jt * 16;
    const float* ebase = erow + (size_t)j0 * 256 + (size_t)n16 * 256 + half * 8;
    v8f c = {0.f, 0.f, 0.f, 0.f, 0.f, 0.f, 0.f, 0.f};
    float ss = 0.f;                      // sum of squares for row j0 + n16 (this lane)
#pragma unroll
    for (int t = 0; t < 8; ++t) {
      const float* pa = (const float*)__builtin_assume_aligned(ebase + t * 32, 16);
      float4 f0 = *(const float4*)pa;            // K = 32t + 8h + 0..3
      float4 f1 = *(const float4*)(pa + 4);      // K = 32t + 8h + 4..7
      float4 f2 = *(const float4*)(pa + 16);     // K = 32t + 8h + 16..19
      float4 f3 = *(const float4*)(pa + 20);     // K = 32t + 8h + 20..23
      ss = fmaf(f0.x, f0.x, ss); ss = fmaf(f0.y, f0.y, ss);
      ss = fmaf(f0.z, f0.z, ss); ss = fmaf(f0.w, f0.w, ss);
      ss = fmaf(f1.x, f1.x, ss); ss = fmaf(f1.y, f1.y, ss);
      ss = fmaf(f1.z, f1.z, ss); ss = fmaf(f1.w, f1.w, ss);
      ss = fmaf(f2.x, f2.x, ss); ss = fmaf(f2.y, f2.y, ss);
      ss = fmaf(f2.z, f2.z, ss); ss = fmaf(f2.w, f2.w, ss);
      ss = fmaf(f3.x, f3.x, ss); ss = fmaf(f3.y, f3.y, ss);
      ss = fmaf(f3.z, f3.z, ss); ss = fmaf(f3.w, f3.w, ss);
      v16bf a;
      a[0]  = (bf16_t)f0.x; a[1]  = (bf16_t)f0.y; a[2]  = (bf16_t)f0.z; a[3]  = (bf16_t)f0.w;
      a[4]  = (bf16_t)f1.x; a[5]  = (bf16_t)f1.y; a[6]  = (bf16_t)f1.z; a[7]  = (bf16_t)f1.w;
      a[8]  = (bf16_t)f2.x; a[9]  = (bf16_t)f2.y; a[10] = (bf16_t)f2.z; a[11] = (bf16_t)f2.w;
      a[12] = (bf16_t)f3.x; a[13] = (bf16_t)f3.y; a[14] = (bf16_t)f3.z; a[15] = (bf16_t)f3.w;
      c = wmma_bf16(a, btile[t], c);
    }
    ss += __shfl_xor(ss, 16);            // combine the two K-halves of row j0+n16

    // epilogue: C[m][n] = dot(row j0+m, head n); scale per row, add b_edge, store
#pragma unroll
    for (int r = 0; r < 8; ++r) {
      int j = j0 + r + 8 * half;                       // C row -> j index
      float ssr = __shfl(ss, r + 8 * half);            // row's sum of squares
      float sc = rsqrtf(ssr * (1.0f / 256.0f) + 1e-5f);
      if (n16 < 8 && j <= i)
        bias_bi[(size_t)n16 * 512 * 512 + j] = c[r] * sc + beL;
    }
  }
}

// ------------------------------------------------------------ attention ----
// One wave per (b,h, 16-row i tile). Flash-style online softmax over causal j.
__global__ __launch_bounds__(256) void attn_kernel(const bf16_t* __restrict__ q_ws,
                                                   const bf16_t* __restrict__ k_ws,
                                                   const bf16_t* __restrict__ vt_ws,
                                                   const float* __restrict__ bias_ws,
                                                   const unsigned char* __restrict__ mask,
                                                   bf16_t* __restrict__ o_bf) {
  __shared__ bf16_t p_lds[8][16 * 32];
  int wavein = threadIdx.x >> 5;
  int wave = blockIdx.x * (blockDim.x >> 5) + wavein;
  int bh = wave >> 5;          // 32 i-tiles per (b,h)
  int itile = wave & 31;
  int bidx = bh >> 3;
  int lane = threadIdx.x & 31, n16 = lane & 15, half = lane >> 4;

  const bf16_t* qp = q_ws + (size_t)bh * 512 * 64;
  const bf16_t* kp = k_ws + (size_t)bh * 512 * 64;
  const bf16_t* vtp = vt_ws + (size_t)bh * 64 * 512;
  const float* bp = bias_ws + (size_t)bh * 512 * 512;
  const unsigned char* mk = mask + bidx * 512;
  bf16_t* pl = &p_lds[wavein][0];

  int i0 = itile * 16;
  v16bf aq0 = load_a_tile(qp + i0 * 64, 64);       // q rows i0..i0+15, d 0..31
  v16bf aq1 = load_a_tile(qp + i0 * 64 + 32, 64);  // d 32..63

  v8f zero = {0.f, 0.f, 0.f, 0.f, 0.f, 0.f, 0.f, 0.f};
  v8f oc[4] = {zero, zero, zero, zero};            // O tile: 16 x 64 (four N=16 tiles)
  float mrun[8], lrun[8];
#pragma unroll
  for (int r = 0; r < 8; ++r) { mrun[r] = NEG_MAX; lrun[r] = 0.f; }

  int jtiles = itile + 1;                          // 16-wide j tiles with any valid j<=i
  for (int jt = 0; jt < jtiles; jt += 2) {
    int j0 = jt * 16;
    bool have2 = (jt + 1) < jtiles;                // wave-uniform

    v8f s0 = zero, s1 = zero;
    s0 = wmma_bf16(aq0, load_bt_tile(kp + j0 * 64, 64), s0);
    s0 = wmma_bf16(aq1, load_bt_tile(kp + j0 * 64 + 32, 64), s0);
    if (have2) {
      s1 = wmma_bf16(aq0, load_bt_tile(kp + (j0 + 16) * 64, 64), s1);
      s1 = wmma_bf16(aq1, load_bt_tile(kp + (j0 + 16) * 64 + 32, 64), s1);
    }

#pragma unroll
    for (int r = 0; r < 8; ++r) {
      int m = r + 8 * half;
      int i = i0 + m;
      int j = j0 + n16;
      int j1 = j0 + 16 + n16;
      float v0 = NEG_MAX, v1 = NEG_MAX;
      if (j <= i && mk[j])               v0 = s0[r] + bp[(size_t)i * 512 + j];
      if (have2 && j1 <= i && mk[j1])    v1 = s1[r] + bp[(size_t)i * 512 + j1];

      float mx = fmaxf(v0, v1);
      mx = fmaxf(mx, __shfl_xor(mx, 1));
      mx = fmaxf(mx, __shfl_xor(mx, 2));
      mx = fmaxf(mx, __shfl_xor(mx, 4));
      mx = fmaxf(mx, __shfl_xor(mx, 8));           // row max within 16-lane group
      float mnew = fmaxf(mrun[r], mx);
      float alpha = __expf(mrun[r] - mnew);
      mrun[r] = mnew;
      float p0 = (v0 > -1e37f) ? __expf(v0 - mnew) : 0.f;
      float p1 = (v1 > -1e37f) ? __expf(v1 - mnew) : 0.f;
      lrun[r] = lrun[r] * alpha + p0 + p1;
      oc[0][r] *= alpha; oc[1][r] *= alpha; oc[2][r] *= alpha; oc[3][r] *= alpha;
      pl[m * 32 + n16]      = (bf16_t)p0;          // stage P (C-layout -> A-layout via LDS)
      pl[m * 32 + 16 + n16] = (bf16_t)p1;
    }

    v16bf ap = load_a_tile(pl, 32);                // P as A-operand (16 x 32)
#pragma unroll
    for (int t = 0; t < 4; ++t)
      oc[t] = wmma_bf16(ap, load_bt_tile(vtp + (t * 16) * 512 + j0, 512), oc[t]);
  }

  int h = bh & 7;
#pragma unroll
  for (int r = 0; r < 8; ++r) {
    float l = lrun[r];
    l += __shfl_xor(l, 1); l += __shfl_xor(l, 2);
    l += __shfl_xor(l, 4); l += __shfl_xor(l, 8);
    float inv = 1.0f / l;
    int row = bidx * 512 + i0 + r + 8 * half;
#pragma unroll
    for (int t = 0; t < 4; ++t)
      o_bf[(size_t)row * 512 + h * 64 + t * 16 + n16] = (bf16_t)(oc[t][r] * inv);
  }
}

// ------------------------------------------------------------ out GEMM ----
// O [1024 x 512] bf16  @  W_out -> out [1024 x 256] f32.  woutT is [256 x 512].
__global__ __launch_bounds__(256) void out_gemm_kernel(const bf16_t* __restrict__ o_bf,
                                                       const bf16_t* __restrict__ woutT,
                                                       float* __restrict__ out) {
  int wave = (blockIdx.x * blockDim.x + threadIdx.x) >> 5;
  int mt = wave >> 4, nt = wave & 15;
  v8f acc = {0.f, 0.f, 0.f, 0.f, 0.f, 0.f, 0.f, 0.f};
#pragma unroll
  for (int k0 = 0; k0 < 512; k0 += 32)
    acc = wmma_bf16(load_a_tile(o_bf + (size_t)mt * 16 * 512 + k0, 512),
                    load_bt_tile(woutT + (size_t)nt * 16 * 512 + k0, 512), acc);
  int lane = threadIdx.x & 31, n16 = lane & 15, half = lane >> 4;
  float* dst = out + (size_t)(mt * 16) * 256 + nt * 16;
#pragma unroll
  for (int r = 0; r < 8; ++r) dst[(r + 8 * half) * 256 + n16] = acc[r];
}

// ------------------------------------------------------------ launcher ----
extern "C" void kernel_launch(void* const* d_in, const int* in_sizes, int n_in,
                              void* d_out, int out_size, void* d_ws, size_t ws_size,
                              hipStream_t stream) {
  const float*         x       = (const float*)d_in[0];
  const unsigned char* mask    = (const unsigned char*)d_in[1];   // jnp bool -> 1 byte
  const float*         edges   = (const float*)d_in[2];
  const float*         gamma_x = (const float*)d_in[3];
  const float*         W_qkv   = (const float*)d_in[4];
  const float*         gamma_e = (const float*)d_in[5];
  const float*         W_edge  = (const float*)d_in[6];
  const float*         b_edge  = (const float*)d_in[7];
  const float*         W_out   = (const float*)d_in[8];
  float* out = (float*)d_out;
  char*  ws  = (char*)d_ws;

  // workspace layout (bytes), 256-aligned
  bf16_t* xn      = (bf16_t*)(ws + 0);          // 1024*256*2      =   524288
  bf16_t* wqkvT   = (bf16_t*)(ws + 524288);     // 1536*256*2      =   786432
  bf16_t* woutT   = (bf16_t*)(ws + 1310720);    // 256*512*2       =   262144
  bf16_t* q_ws    = (bf16_t*)(ws + 1572864);    // 16*512*64*2     =  1048576
  bf16_t* k_ws    = (bf16_t*)(ws + 2621440);    //                 =  1048576
  bf16_t* vt_ws   = (bf16_t*)(ws + 3670016);    //                 =  1048576
  bf16_t* o_bf    = (bf16_t*)(ws + 4718592);    // 1024*512*2      =  1048576
  float*  bias_ws = (float*)(ws + 5767168);     // 2*8*512*512*4   = 16777216
  bf16_t* gwT     = (bf16_t*)(ws + 22544384);   // 16*256*2        =     8192

  rmsnorm_x_kernel<<<1024, 256, 0, stream>>>(x, gamma_x, xn);
  transpose_to_bf16<<<(256 * 1536 + 255) / 256, 256, 0, stream>>>(W_qkv, wqkvT, 256, 1536);
  transpose_to_bf16<<<(512 * 256 + 255) / 256, 256, 0, stream>>>(W_out, woutT, 512, 256);
  prep_gwT_kernel<<<16, 256, 0, stream>>>(gamma_e, W_edge, gwT);
  qkv_gemm_kernel<<<768, 256, 0, stream>>>(xn, wqkvT, q_ws, k_ws, vt_ws);
  edge_bias_kernel<<<1024, 256, 0, stream>>>(edges, gwT, b_edge, bias_ws);
  attn_kernel<<<64, 256, 0, stream>>>(q_ws, k_ws, vt_ws, bias_ws, mask, o_bf);
  out_gemm_kernel<<<128, 256, 0, stream>>>(o_bf, woutT, out);
}
